// BTLMAttention_36180804501899
// MI455X (gfx1250) — matrix operand order repaired and checked by reading, not compiled, not run on hardware
//
#include <hip/hip_runtime.h>
#include <math.h>

typedef __attribute__((ext_vector_type(16))) __bf16 v16bf;
typedef __attribute__((ext_vector_type(8)))  __bf16 bf16x8;
typedef __attribute__((ext_vector_type(8)))  float  v8f;

#define S_LEN 2048
#define HID   2560
#define NH    32
#define HD    80
#define HDP   96          // head dim padded to 3*32 for K=32 WMMA tiling
#define NEGV  (-1e9f)

#if defined(__has_builtin)
#if __has_builtin(__builtin_amdgcn_global_load_async_to_lds_b128)
#define HAVE_ASYNC_LDS 1
#endif
#endif

#ifdef HAVE_ASYNC_LDS
typedef __attribute__((__vector_size__(4 * sizeof(int)))) int v4i_;
typedef __attribute__((address_space(1))) v4i_ gv4i;
typedef __attribute__((address_space(3))) v4i_ lv4i;
__device__ __forceinline__ void async_b128(const void* g, void* l) {
  // integer->pointer casts build the AS-qualified v4i pointers the builtin
  // expects; low 32 bits of a generic LDS address == LDS offset (ISA aperture).
  __builtin_amdgcn_global_load_async_to_lds_b128(
      (gv4i*)(unsigned long long)(uintptr_t)g,
      (lv4i*)(unsigned int)(uintptr_t)l, 0, 0);
}
#define ASYNC_WAIT() asm volatile("s_wait_asynccnt 0x0" ::: "memory")
#else
#define ASYNC_WAIT()
#endif

__device__ __forceinline__ void copy_b128(const void* g, void* l) {
#ifdef HAVE_ASYNC_LDS
  async_b128(g, l);
#else
  *(bf16x8*)l = *(const bf16x8*)g;
#endif
}

__device__ __forceinline__ v16bf cat8(bf16x8 lo, bf16x8 hi) {
  return __builtin_shufflevector(lo, hi, 0, 1, 2, 3, 4, 5, 6, 7,
                                 8, 9, 10, 11, 12, 13, 14, 15);
}

// ---------------------------------------------------------------------------
// fp32 -> bf16 elementwise convert (grid-stride)
// ---------------------------------------------------------------------------
__global__ void f32_to_bf16_kernel(const float* __restrict__ in,
                                   __bf16* __restrict__ out, int n) {
  for (int i = blockIdx.x * blockDim.x + threadIdx.x; i < n;
       i += gridDim.x * blockDim.x)
    out[i] = (__bf16)in[i];
}

// ---------------------------------------------------------------------------
// fp32 [R][C] -> bf16 TRANSPOSED [C][R], tiled 16x16 via LDS (coalesced both
// sides). Weights are pre-transposed so GEMM B-tiles stage as contiguous b128.
// ---------------------------------------------------------------------------
__global__ __launch_bounds__(256) void f32_to_bf16t_kernel(
    const float* __restrict__ in, __bf16* __restrict__ out, int R, int C) {
  __shared__ float t[16][17];
  const int r0 = blockIdx.y * 16, c0 = blockIdx.x * 16;
  const int r = threadIdx.x >> 4, c = threadIdx.x & 15;
  t[c][r] = in[(size_t)(r0 + r) * C + c0 + c];
  __syncthreads();
  out[(size_t)(c0 + r) * R + r0 + c] = (__bf16)t[r][c];
}

// ---------------------------------------------------------------------------
// bf16 [NH][S][HD] -> bf16 [NH][HD][S] (per-head V transpose), tiled 16x16.
// ---------------------------------------------------------------------------
__global__ __launch_bounds__(256) void transpose_v_kernel(
    const __bf16* __restrict__ Vin, __bf16* __restrict__ Vout) {
  const int h  = blockIdx.z;
  const int s0 = blockIdx.x * 16;
  const int d0 = blockIdx.y * 16;
  __shared__ __bf16 t[16][17];
  const int r = threadIdx.x >> 4, c = threadIdx.x & 15;
  t[c][r] = Vin[((size_t)h * S_LEN + s0 + r) * HD + d0 + c];
  __syncthreads();
  Vout[((size_t)h * HD + d0 + r) * S_LEN + s0 + c] = t[r][c];
}

// ---------------------------------------------------------------------------
// bf16 GEMM  C[M,N] = A[M,K] * Bt[N,K]^T, fp32 out.
// 64x64 workgroup tile, 8 waves (4x2), each wave -> 16x32 via two WMMA accs.
// DOUBLE-BUFFERED async global->LDS staging: tile k+1 streams in while tile k
// is consumed by the matrix units; one barrier + one asynccnt wait per step.
// ---------------------------------------------------------------------------
__global__ __launch_bounds__(256) void gemm_bf16_kernel(
    const __bf16* __restrict__ A, const __bf16* __restrict__ Bt,
    float* __restrict__ C, int M, int N, int K) {
  __shared__ alignas(16) __bf16 sA[2][64][32];    // [buf][m][k]
  __shared__ alignas(16) __bf16 sBt[2][64][32];   // [buf][n][k]

  const int m0   = blockIdx.y * 64;
  const int n0   = blockIdx.x * 64;
  const int wave = threadIdx.x >> 5;
  const int lane = threadIdx.x & 31;
  const int ln   = lane & 15;
  const int half = lane >> 4;
  const int wm   = (wave >> 1) * 16;   // 0,16,32,48
  const int wn   = (wave & 1) * 32;    // 0,32

  // staging coordinates (one 16B chunk per thread per tile)
  const int ra = threadIdx.x >> 2;         // tile row (4 chunks/row)
  const int ca = (threadIdx.x & 3) * 8;    // tile col

  auto stage = [&](int buf, int kk) {
    copy_b128(&A[(size_t)(m0 + ra) * K + kk + ca], &sA[buf][ra][ca]);
    copy_b128(&Bt[(size_t)(n0 + ra) * K + kk + ca], &sBt[buf][ra][ca]);
  };

  v8f acc0 = {};
  v8f acc1 = {};

  stage(0, 0);
  ASYNC_WAIT();
  __syncthreads();

  for (int k0 = 0; k0 < K; k0 += 32) {
    const int cur = (k0 >> 5) & 1;
    const int nxt = cur ^ 1;
    if (k0 + 32 < K) {
      stage(nxt, k0 + 32);  // async issue only; latency hidden under WMMAs
      __builtin_prefetch(&A[(size_t)(m0 + (threadIdx.x & 63)) * K + k0 + 64],
                         0, 1);
    }

    // A fragment: contiguous along k: k = half*16 .. half*16+15
    v16bf a = cat8(*(const bf16x8*)&sA[cur][wm + ln][half * 16],
                   *(const bf16x8*)&sA[cur][wm + ln][half * 16 + 8]);
    // B fragments: contiguous along k in sBt: k = half*16 .. half*16+15
    v16bf b0 = cat8(*(const bf16x8*)&sBt[cur][wn + ln][half * 16],
                    *(const bf16x8*)&sBt[cur][wn + ln][half * 16 + 8]);
    v16bf b1 = cat8(*(const bf16x8*)&sBt[cur][wn + 16 + ln][half * 16],
                    *(const bf16x8*)&sBt[cur][wn + 16 + ln][half * 16 + 8]);
    acc0 = __builtin_amdgcn_wmma_f32_16x16x32_bf16(false, a, false, b0,
                                                   (short)0, acc0, false, false);
    acc1 = __builtin_amdgcn_wmma_f32_16x16x32_bf16(false, a, false, b1,
                                                   (short)0, acc1, false, false);

    ASYNC_WAIT();       // next tile resident
    __syncthreads();    // single barrier per k-step
  }

#pragma unroll
  for (int r = 0; r < 8; ++r) {
    int row = m0 + wm + r + half * 8;  // C/D layout: M = r + half*8
    C[(size_t)row * N + n0 + wn + ln]      = acc0[r];
    C[(size_t)row * N + n0 + wn + 16 + ln] = acc1[r];
  }
}

// ---------------------------------------------------------------------------
// RoPE (+SCALE folded into Q) and repack to head-major bf16:
//   Q,K: [NH][S][HDP] (zero-padded 80..95), V: [NH][S][HD]
// ---------------------------------------------------------------------------
__global__ __launch_bounds__(HDP) void rope_pack_kernel(
    const float* __restrict__ qkv, const int* __restrict__ pos_ids,
    __bf16* __restrict__ Qo, __bf16* __restrict__ Ko, __bf16* __restrict__ Vo) {
  const int s = blockIdx.x;
  const int h = blockIdx.y;
  const int d = threadIdx.x;      // 0..95
  const float scale = 0.11180339887498949f;  // 80^-0.5

  float qv = 0.f, kv = 0.f;
  const size_t base = (size_t)s * (3 * HID) + h * HD;
  if (d < HD) {
    qv = qkv[base + d];
    kv = qkv[base + HID + d];
  }
  if (d < 64) {
    int i  = d & 31;
    int hi = d >> 5;
    float inv = __expf(-(2.f * (float)i / 64.f) * 9.210340371976184f);
    float ang = (float)pos_ids[s] * inv;
    float c = __cosf(ang), sn = __sinf(ang);
    float qp = qkv[base + (hi ? d - 32 : d + 32)];
    float kp = qkv[base + HID + (hi ? d - 32 : d + 32)];
    if (!hi) { qv = qv * c - qp * sn; kv = kv * c - kp * sn; }
    else     { qv = qv * c + qp * sn; kv = kv * c + kp * sn; }
  }
  const size_t ob = ((size_t)h * S_LEN + s) * HDP + d;
  Qo[ob] = (__bf16)((d < HD) ? qv * scale : 0.f);
  Ko[ob] = (__bf16)((d < HD) ? kv : 0.f);
  if (d < HD)
    Vo[((size_t)h * S_LEN + s) * HD + d] = (__bf16)qkv[base + 2 * HID + d];
}

// ---------------------------------------------------------------------------
// Flash attention with ALiBi + causal. One workgroup = (head, 64 q-rows),
// 4 waves x 16 q-rows. K and pre-transposed V tiles DOUBLE-BUFFERED via async
// b128 copies; all hot-loop fragment reads are ds_load_b128.
// ---------------------------------------------------------------------------
__global__ __launch_bounds__(128) void attn_kernel(
    const __bf16* __restrict__ Q, const __bf16* __restrict__ Kc,
    const __bf16* __restrict__ Vt, __bf16* __restrict__ O) {
  const int h    = blockIdx.x;
  const int qblk = blockIdx.y;
  const int wave = threadIdx.x >> 5;
  const int lane = threadIdx.x & 31;
  const int ln   = lane & 15;
  const int half = lane >> 4;
  const int q0   = qblk * 64 + wave * 16;
  const float slope = exp2f(-8.f * (float)(h + 1) / (float)NH);

  __shared__ alignas(16) __bf16 sK[2][16][HDP];   // [buf][key][d]
  __shared__ alignas(16) __bf16 sVt[2][HD][16];   // [buf][d][key]
  __shared__ alignas(16) __bf16 sPb[4][16][16];   // prob relay, bf16

  const __bf16* Qh  = Q  + (size_t)h * S_LEN * HDP;
  const __bf16* Kh  = Kc + (size_t)h * S_LEN * HDP;
  const __bf16* Vth = Vt + (size_t)h * HD * S_LEN;

  auto stageKV = [&](int buf, int kb_) {
    const __bf16* gk = Kh + (size_t)(kb_ * 16) * HDP;
    __bf16* lk = &sK[buf][0][0];
    for (int c = threadIdx.x; c < 192; c += 128)      // K tile 16x96
      copy_b128(gk + c * 8, lk + c * 8);
    for (int c = threadIdx.x; c < 160; c += 128) {    // V^T tile 80x16
      int d = c >> 1, g = (c & 1) * 8;
      copy_b128(&Vth[(size_t)d * S_LEN + kb_ * 16 + g], &sVt[buf][d][g]);
    }
  };

  // Q A-fragments (K = 96 -> 3 frags), two b128 global loads each, resident.
  v16bf qa[3];
#pragma unroll
  for (int f = 0; f < 3; ++f) {
    const __bf16* qrow = &Qh[(size_t)(q0 + ln) * HDP + f * 32];
    qa[f] = cat8(*(const bf16x8*)&qrow[half * 16],
                 *(const bf16x8*)&qrow[half * 16 + 8]);
  }

  bf16x8 z8 = {};
  v8f zero8 = {};
  v8f accO[5];
#pragma unroll
  for (int t = 0; t < 5; ++t) accO[t] = zero8;
  float mrun[8], lrun[8];
#pragma unroll
  for (int r = 0; r < 8; ++r) { mrun[r] = -1e30f; lrun[r] = 0.f; }

  const int kbmax = qblk * 4 + 3;  // causal upper bound for this workgroup
  stageKV(0, 0);
  ASYNC_WAIT();
  __syncthreads();

  for (int kb = 0; kb <= kbmax; ++kb) {
    const int cur = kb & 1;
    const int nxt = cur ^ 1;
    if (kb < kbmax) stageKV(nxt, kb + 1);  // overlap with compute below

    if (kb * 16 <= q0 + 15) {  // wave-uniform causal skip
      // scores = Q * K^T ; B frag contiguous along d in sK
      v8f sc = zero8;
#pragma unroll
      for (int f = 0; f < 3; ++f) {
        const __bf16* krow = &sK[cur][ln][f * 32 + half * 16];
        v16bf bk = cat8(*(const bf16x8*)&krow[0], *(const bf16x8*)&krow[8]);
        sc = __builtin_amdgcn_wmma_f32_16x16x32_bf16(false, qa[f], false, bk,
                                                     (short)0, sc, false, false);
      }
      // ALiBi + causal mask + online softmax (row stats via 16-lane shfl_xor)
#pragma unroll
      for (int r = 0; r < 8; ++r) {
        int qrow = q0 + r + half * 8;
        int kpos = kb * 16 + ln;
        float rel = (float)(kpos - qrow);
        float sv = (rel <= 0.f) ? (sc[r] + slope * rel) : NEGV;
        float mx = sv;
        mx = fmaxf(mx, __shfl_xor(mx, 1));
        mx = fmaxf(mx, __shfl_xor(mx, 2));
        mx = fmaxf(mx, __shfl_xor(mx, 4));
        mx = fmaxf(mx, __shfl_xor(mx, 8));
        float mnew = fmaxf(mrun[r], mx);
        float pv = __expf(sv - mnew);
        float sum = pv;
        sum += __shfl_xor(sum, 1);
        sum += __shfl_xor(sum, 2);
        sum += __shfl_xor(sum, 4);
        sum += __shfl_xor(sum, 8);
        float corr = __expf(mrun[r] - mnew);
        lrun[r] = lrun[r] * corr + sum;
        mrun[r] = mnew;
#pragma unroll
        for (int t = 0; t < 5; ++t) accO[t][r] = accO[t][r] * corr;
        sPb[wave][r + half * 8][ln] = (__bf16)pv;  // C-layout -> A-layout relay
      }
      asm volatile("s_wait_dscnt 0x0" ::: "memory");  // same-wave LDS RAW

      // P A-fragment: half 0 holds k=0..15 (two b128), half 1 zero-padded
      v16bf pa = (half == 0) ? cat8(*(const bf16x8*)&sPb[wave][ln][0], *(const bf16x8*)&sPb[wave][ln][8]) : cat8(z8, z8);

      // accO += P * V ; V B-frag contiguous along key in sVt (half==1 -> 0)
#pragma unroll
      for (int t = 0; t < 5; ++t) {
        bf16x8 vlo = z8, vhi = z8;
        if (half == 0) {
          vlo = *(const bf16x8*)&sVt[cur][t * 16 + ln][0];
          vhi = *(const bf16x8*)&sVt[cur][t * 16 + ln][8];
        }
        v16bf vb = cat8(vlo, vhi);
        accO[t] = __builtin_amdgcn_wmma_f32_16x16x32_bf16(
            false, pa, false, vb, (short)0, accO[t], false, false);
      }
    }
    ASYNC_WAIT();       // next K/V tile resident
    __syncthreads();    // single barrier per k-block
  }

  // normalize and write attention output as bf16 [S][HID] for out-projection
#pragma unroll
  for (int r = 0; r < 8; ++r) {
    float inv = 1.f / lrun[r];
    int row = q0 + r + half * 8;
#pragma unroll
    for (int t = 0; t < 5; ++t)
      O[(size_t)row * HID + h * HD + t * 16 + ln] = (__bf16)(accO[t][r] * inv);
  }
}

// ---------------------------------------------------------------------------
extern "C" void kernel_launch(void* const* d_in, const int* in_sizes, int n_in,
                              void* d_out, int out_size, void* d_ws,
                              size_t ws_size, hipStream_t stream) {
  (void)in_sizes; (void)n_in; (void)out_size; (void)ws_size;
  const int*   pos    = (const int*)d_in[0];
  const float* hidden = (const float*)d_in[1];
  const float* w_qkv  = (const float*)d_in[2];
  const float* w_out  = (const float*)d_in[3];
  float*       out    = (float*)d_out;

  char* ws = (char*)d_ws;
  size_t off = 0;
  auto carve = [&](size_t bytes) {
    void* p = ws + off;
    off = (off + bytes + 255) & ~(size_t)255;
    return p;
  };
  __bf16* hsb    = (__bf16*)carve((size_t)S_LEN * HID * 2);
  __bf16* wqkvbt = (__bf16*)carve((size_t)HID * 3 * HID * 2);  // [3H][H]
  __bf16* woutbt = (__bf16*)carve((size_t)HID * HID * 2);      // [H][H]
  float*  qkv    = (float*)carve((size_t)S_LEN * 3 * HID * 4);
  __bf16* qb     = (__bf16*)carve((size_t)NH * S_LEN * HDP * 2);
  __bf16* kb     = (__bf16*)carve((size_t)NH * S_LEN * HDP * 2);
  __bf16* vb     = (__bf16*)carve((size_t)NH * S_LEN * HD * 2);
  __bf16* vtb    = (__bf16*)carve((size_t)NH * HD * S_LEN * 2);
  __bf16* attnb  = (__bf16*)carve((size_t)S_LEN * HID * 2);

  f32_to_bf16_kernel<<<2048, 256, 0, stream>>>(hidden, hsb, S_LEN * HID);
  // weights converted AND transposed -> GEMM B-tiles stage contiguously
  dim3 tq(3 * HID / 16, HID / 16);
  f32_to_bf16t_kernel<<<tq, 256, 0, stream>>>(w_qkv, wqkvbt, HID, 3 * HID);
  dim3 to(HID / 16, HID / 16);
  f32_to_bf16t_kernel<<<to, 256, 0, stream>>>(w_out, woutbt, HID, HID);

  dim3 g1(3 * HID / 64, S_LEN / 64);
  gemm_bf16_kernel<<<g1, 256, 0, stream>>>(hsb, wqkvbt, qkv, S_LEN, 3 * HID, HID);

  dim3 g2(S_LEN, NH);
  rope_pack_kernel<<<g2, HDP, 0, stream>>>(qkv, pos, qb, kb, vb);

  dim3 gt(S_LEN / 16, HD / 16, NH);
  transpose_v_kernel<<<gt, 256, 0, stream>>>(vb, vtb);

  dim3 g3(NH, S_LEN / 64);
  attn_kernel<<<g3, 128, 0, stream>>>(qb, kb, vtb, attnb);

  dim3 g4(HID / 64, S_LEN / 64);
  gemm_bf16_kernel<<<g4, 256, 0, stream>>>(attnb, woutbt, out, S_LEN, HID, HID);
}
